// RelativePositionalEncoding_8169027797140
// MI455X (gfx1250) — matrix operand order repaired and checked by reading, not compiled, not run on hardware
//
#include <hip/hip_runtime.h>

#define S_LEN 2048
#define DMODEL 1024
#define NHEADS 16
#define DHEAD 64
#define BATCH 2

typedef _Float16 half_t;
typedef __attribute__((ext_vector_type(16))) _Float16 v16h;
typedef __attribute__((ext_vector_type(8)))  _Float16 v8h;
typedef __attribute__((ext_vector_type(4)))  _Float16 v4h;
typedef __attribute__((ext_vector_type(8)))  float    v8f;
typedef __attribute__((ext_vector_type(4)))  float    v4f;

__device__ __forceinline__ int imin(int a, int b) { return a < b ? a : b; }
__device__ __forceinline__ int imax(int a, int b) { return a > b ? a : b; }

// ---------------------------------------------------------------------------
// WMMA helper: D = A(16x32 f16) x B(32x16 f16) + C(16x16 f32)
// ---------------------------------------------------------------------------
__device__ __forceinline__ v8f wmma_f16(v16h a, v16h b, v8f c) {
    return __builtin_amdgcn_wmma_f32_16x16x32_f16(
        /*neg_a=*/false, a, /*neg_b=*/false, b,
        /*c_mod=*/(short)0, c, /*reuse_a=*/false, /*reuse_b=*/false);
}

// A-fragment (16x32, M x K), row-major source. Lane l holds row M=l&15.
// Element e -> K = e + 16h, h = l>>4.
// => two contiguous 8xf16 (16B) runs at K = 16h and K = 16h+8.
__device__ __forceinline__ v16h load_a16(const half_t* row, int h) {
    const v8h lo = *(const v8h*)(row + 16 * h);
    const v8h hi = *(const v8h*)(row + 16 * h + 8);
    v16h r;
#pragma unroll
    for (int i = 0; i < 8; ++i) { r[i] = lo[i]; r[i + 8] = hi[i]; }
    return r;
}
// Same, but with independently clamped run start offsets (for causal edge).
__device__ __forceinline__ v16h load_a16_runs(const half_t* row, int k1, int k2) {
    const v8h lo = *(const v8h*)(row + k1);
    const v8h hi = *(const v8h*)(row + k2);
    v16h r;
#pragma unroll
    for (int i = 0; i < 8; ++i) { r[i] = lo[i]; r[i + 8] = hi[i]; }
    return r;
}

// B-fragment (32x16, K x N), source row-major in [N][K] (transposed) layout.
// Lane l holds column N=l&15; element e -> K = e + 16h: one contiguous 16xf16 run.
__device__ __forceinline__ v16h load_b16(const half_t* row, int h) {
    const v8h lo = *(const v8h*)(row + 16 * h);
    const v8h hi = *(const v8h*)(row + 16 * h + 8);
    v16h r;
#pragma unroll
    for (int i = 0; i < 8; ++i) { r[i] = lo[i]; r[i + 8] = hi[i]; }
    return r;
}

// ---------------------------------------------------------------------------
// Pre-pass: f32 -> f16 elementwise (n % 4 == 0)
// ---------------------------------------------------------------------------
__global__ void cvt_f32_f16_v4(const float* __restrict__ in, half_t* __restrict__ out, int n) {
    int i = (blockIdx.x * blockDim.x + threadIdx.x) * 4;
    int stride = gridDim.x * blockDim.x * 4;
    for (; i < n; i += stride) {
        v4f v = *(const v4f*)(in + i);
        v4h o;
        o[0] = (half_t)v[0]; o[1] = (half_t)v[1]; o[2] = (half_t)v[2]; o[3] = (half_t)v[3];
        *(v4h*)(out + i) = o;
    }
}

// Pre-pass: transpose + convert W[d][j] (f32) -> Wt[j][d] (f16), 3 matrices.
__global__ void transpose_cvt_w(const float* __restrict__ Wq, const float* __restrict__ Wk,
                                const float* __restrict__ Wv, half_t* __restrict__ Wt) {
    __shared__ float tile[32][33];
    const float* W = blockIdx.z == 0 ? Wq : (blockIdx.z == 1 ? Wk : Wv);
    half_t* out = Wt + (size_t)blockIdx.z * DMODEL * DMODEL;
    const int d0 = blockIdx.y * 32, j0 = blockIdx.x * 32;
#pragma unroll
    for (int i = 0; i < 32; i += 8)
        tile[threadIdx.y + i][threadIdx.x] =
            W[(size_t)(d0 + threadIdx.y + i) * DMODEL + j0 + threadIdx.x];
    __syncthreads();
#pragma unroll
    for (int i = 0; i < 32; i += 8)
        out[(size_t)(j0 + threadIdx.y + i) * DMODEL + d0 + threadIdx.x] =
            (half_t)tile[threadIdx.x][threadIdx.y + i];
}

// ---------------------------------------------------------------------------
// QKV projection: q/k = x @ W + b  -> f16 [b,h,s,dh];  v -> transposed [b,h,dh,s]
// One wave: 32 rows x 64 cols of one of {q,k,v}: 2 A-frags reused across
// 4 B-frags => 8 WMMAs per 12 b128 loads per K-step. blockDim=128 (4 waves).
// grid = (B*S/32/4, D/64, 3)
// ---------------------------------------------------------------------------
__global__ void __launch_bounds__(128) qkv_gemm(
    const half_t* __restrict__ x16, const half_t* __restrict__ Wt16,
    const float* __restrict__ bq, const float* __restrict__ bk, const float* __restrict__ bv,
    half_t* __restrict__ q16, half_t* __restrict__ k16, half_t* __restrict__ vT16) {
    const int lane = threadIdx.x & 31;
    const int w = threadIdx.x >> 5;
    const int nrow = lane & 15, h = lane >> 4;
    const int r0 = (blockIdx.x * 4 + w) * 32;   // global row in [0, B*S)
    const int j0 = blockIdx.y * 64;
    const int m = blockIdx.z;
    const half_t* W = Wt16 + (size_t)m * DMODEL * DMODEL;
    const float* bias = m == 0 ? bq : (m == 1 ? bk : bv);

    v8f acc[2][4] = {};
    const half_t* arow0 = x16 + (size_t)(r0 + nrow) * DMODEL;
    const half_t* arow1 = x16 + (size_t)(r0 + 16 + nrow) * DMODEL;
#pragma unroll 2
    for (int k0 = 0; k0 < DMODEL; k0 += 32) {
        v16h a0 = load_a16(arow0 + k0, h);
        v16h a1 = load_a16(arow1 + k0, h);
#pragma unroll
        for (int t = 0; t < 4; ++t) {
            const half_t* brow = W + (size_t)(j0 + t * 16 + nrow) * DMODEL;
            v16h bfr = load_b16(brow + k0, h);
            acc[0][t] = wmma_f16(a0, bfr, acc[0][t]);
            acc[1][t] = wmma_f16(a1, bfr, acc[1][t]);
        }
    }
    const int bb = r0 >> 11;              // batch (S_LEN = 2048, r0 % 32 == 0)
#pragma unroll
    for (int g = 0; g < 2; ++g) {
        const int srow0 = (r0 & (S_LEN - 1)) + g * 16;
#pragma unroll
        for (int t = 0; t < 4; ++t) {
            const int j = j0 + t * 16 + nrow;   // output column for this lane (C: N=lane&15)
            const int hidx = j >> 6, jl = j & 63;
            const float bj = bias[j];
            if (m < 2) {
                half_t* outp = (m == 0 ? q16 : k16) +
                               (size_t)(bb * NHEADS + hidx) * S_LEN * DHEAD + jl;
#pragma unroll
                for (int r = 0; r < 8; ++r) {
                    const int s = srow0 + r + 8 * h;   // C: M = r + 8h
                    outp[(size_t)s * DHEAD] = (half_t)(acc[g][t][r] + bj);
                }
            } else {
                v8h pk;
#pragma unroll
                for (int r = 0; r < 8; ++r) pk[r] = (half_t)(acc[g][t][r] + bj);
                const int s = srow0 + 8 * h;
                *(v8h*)(vT16 + ((size_t)(bb * NHEADS + hidx) * DHEAD + jl) * S_LEN + s) = pk;
            }
        }
    }
}

// ---------------------------------------------------------------------------
// Flash attention with relative positional bias (skew folded into a banded GEMM).
//   S^T[t,s] = k_t . q_s              (A = k tile, B = Q^T frags)
//   RT[c,s]  = Er[l0+c] . q_s         (A = Er rows, same B)  c in [0,48)
//   BiasT[t,s] = RT[(t-t0)+15-(s-s0), s]   gathered via per-wave LDS scratch
//   O^T[d,s] += V^T[d,t] P^T[t,s]     (A = vT tile, B = repacked P^T)
// One wave per 16-query tile; blockDim=128, grid=(32, B*H).
// ---------------------------------------------------------------------------
__global__ void __launch_bounds__(128) rel_attn(
    const half_t* __restrict__ q16, const half_t* __restrict__ k16,
    const half_t* __restrict__ vT16, const half_t* __restrict__ er16,
    float* __restrict__ out) {
    __shared__ float ldsR[4][48 * 17];
    const int lane = threadIdx.x & 31;
    const int w = threadIdx.x >> 5;
    const int nrow = lane & 15, h = lane >> 4;
    const int bh = blockIdx.y;
    const int b = bh >> 4, hd = bh & 15;
    const int s0 = (blockIdx.x + 32 * w) * 16;   // interleaved for load balance
    float* L = ldsR[w];

    // Q^T B-fragments (K=d), reused by both S^T and RT GEMMs.
    const half_t* qrow = q16 + ((size_t)bh * S_LEN + s0 + nrow) * DHEAD;
    const v16h qf0 = load_b16(qrow, h);        // d in [0,32)
    const v16h qf1 = load_b16(qrow + 32, h);   // d in [32,64)

    v8f o[4] = {};
    float mrun = -__builtin_inff(), lrun = 0.f;
    const int nchunk = (s0 + 16 + 31) >> 5;
    for (int ch = 0; ch < nchunk; ++ch) {
        const int t0 = ch * 32;
        // Prefetch next chunk's K rows / V^T runs into cache (global_prefetch_b8).
        if (ch + 1 < nchunk) {
            const int tn = imin(t0 + 32 + nrow, S_LEN - 1);
            __builtin_prefetch(k16 + ((size_t)bh * S_LEN + tn) * DHEAD, 0, 3);
            __builtin_prefetch(
                vT16 + ((size_t)bh * DHEAD + nrow * 4) * S_LEN + imin(t0 + 32, S_LEN - 8), 0, 3);
        }
        // --- S^T tiles (two 16-key subtiles) ---
        v8f st[2];
#pragma unroll
        for (int sub = 0; sub < 2; ++sub) {
            const int tr = imin(t0 + sub * 16 + nrow, S_LEN - 1);
            const half_t* kr = k16 + ((size_t)bh * S_LEN + tr) * DHEAD;
            v8f cc = {};
            cc = wmma_f16(load_a16(kr, h), qf0, cc);
            cc = wmma_f16(load_a16(kr + 32, h), qf1, cc);
            st[sub] = cc;
        }
        // --- relative-bias band RT[c,s] = Er[l0+c].q_s, c in [0,48) ---
        const int l0 = S_LEN - 16 - s0 + t0;
#pragma unroll
        for (int ct = 0; ct < 3; ++ct) {
            const int er = imin(imax(l0 + ct * 16 + nrow, 0), S_LEN - 1);
            const half_t* erow = er16 + (size_t)er * DHEAD;
            v8f rc = {};
            rc = wmma_f16(load_a16(erow, h), qf0, rc);
            rc = wmma_f16(load_a16(erow + 32, h), qf1, rc);
#pragma unroll
            for (int r = 0; r < 8; ++r)
                L[(ct * 16 + r + 8 * h) * 17 + nrow] = rc[r];
        }
        // --- gather skewed bias, scale, causal mask ---
#pragma unroll
        for (int sub = 0; sub < 2; ++sub)
#pragma unroll
            for (int r = 0; r < 8; ++r) {
                const int tl = sub * 16 + r + 8 * h;          // t - t0
                const float bias = L[(tl + 15 - nrow) * 17 + nrow];
                const float v = (st[sub][r] + bias) * 0.125f; // 1/sqrt(dh)
                st[sub][r] = (t0 + tl <= s0 + nrow) ? v : -__builtin_inff();
            }
        // --- online softmax (row = query s = lane&15; halves hold t 0-7 / 8-15) ---
        float mx = -__builtin_inff();
#pragma unroll
        for (int sub = 0; sub < 2; ++sub)
#pragma unroll
            for (int r = 0; r < 8; ++r) mx = fmaxf(mx, st[sub][r]);
        mx = fmaxf(mx, __shfl_xor(mx, 16, 32));
        const float mnew = fmaxf(mrun, mx);
        const float alpha = __expf(mrun - mnew);
        float ps = 0.f;
#pragma unroll
        for (int sub = 0; sub < 2; ++sub)
#pragma unroll
            for (int r = 0; r < 8; ++r) {
                const float p = __expf(st[sub][r] - mnew);
                st[sub][r] = p;
                ps += p;
            }
        ps += __shfl_xor(ps, 16, 32);
        lrun = lrun * alpha + ps;
        mrun = mnew;
#pragma unroll
        for (int dt = 0; dt < 4; ++dt)
#pragma unroll
            for (int r = 0; r < 8; ++r) o[dt][r] *= alpha;
        // --- repack P^T (two f32 C tiles) into one f16 B fragment (K=t, 32) ---
        v16h pb;
#pragma unroll
        for (int e = 0; e < 8; ++e) {
            const float topx = __shfl_xor(st[0][e], 16, 32);
            const float botx = __shfl_xor(st[1][e], 16, 32);
            pb[e]     = (half_t)(h ? botx : st[0][e]);
            pb[e + 8] = (half_t)(h ? st[1][e] : topx);
        }
        // --- O^T += V^T x P^T ---
        const int k1 = imin(t0 + 16 * h, S_LEN - 8);
        const int k2 = imin(t0 + 16 * h + 8, S_LEN - 8);
#pragma unroll
        for (int dt = 0; dt < 4; ++dt) {
            const half_t* vr = vT16 + ((size_t)bh * DHEAD + dt * 16 + nrow) * S_LEN;
            o[dt] = wmma_f16(load_a16_runs(vr, k1, k2), pb, o[dt]);
        }
    }
    // --- normalize + store: out[b, s, h*64 + d] fp32, 16B-vector stores ---
    const float inv = 1.f / lrun;
    float* ob = out + (size_t)(b * S_LEN + s0 + nrow) * DMODEL + hd * DHEAD + 8 * h;
#pragma unroll
    for (int dt = 0; dt < 4; ++dt) {
        v4f lo = {o[dt][0] * inv, o[dt][1] * inv, o[dt][2] * inv, o[dt][3] * inv};
        v4f hi = {o[dt][4] * inv, o[dt][5] * inv, o[dt][6] * inv, o[dt][7] * inv};
        *(v4f*)(ob + dt * 16) = lo;
        *(v4f*)(ob + dt * 16 + 4) = hi;
    }
}

// ---------------------------------------------------------------------------
// Workspace layout (bytes): x16 8MB | Wt16 6MB | Er16 256KB | q16 8MB | k16 8MB | vT16 8MB
// ---------------------------------------------------------------------------
extern "C" void kernel_launch(void* const* d_in, const int* in_sizes, int n_in,
                              void* d_out, int out_size, void* d_ws, size_t ws_size,
                              hipStream_t stream) {
    (void)in_sizes; (void)n_in; (void)out_size; (void)ws_size;
    const float* x  = (const float*)d_in[0];
    const float* Wq = (const float*)d_in[1];
    const float* bq = (const float*)d_in[2];
    const float* Wk = (const float*)d_in[3];
    const float* bk = (const float*)d_in[4];
    const float* Wv = (const float*)d_in[5];
    const float* bv = (const float*)d_in[6];
    const float* Er = (const float*)d_in[7];
    char* ws = (char*)d_ws;
    half_t* x16  = (half_t*)(ws);
    half_t* Wt16 = (half_t*)(ws + 8388608);
    half_t* Er16 = (half_t*)(ws + 14680064);
    half_t* q16  = (half_t*)(ws + 14942208);
    half_t* k16  = (half_t*)(ws + 23330816);
    half_t* vT16 = (half_t*)(ws + 31719424);
    float* out = (float*)d_out;

    cvt_f32_f16_v4<<<dim3(1024), dim3(256), 0, stream>>>(x, x16, BATCH * S_LEN * DMODEL);
    cvt_f32_f16_v4<<<dim3(64), dim3(256), 0, stream>>>(Er, Er16, S_LEN * DHEAD);
    transpose_cvt_w<<<dim3(32, 32, 3), dim3(32, 8), 0, stream>>>(Wq, Wk, Wv, Wt16);
    qkv_gemm<<<dim3(32, 16, 3), dim3(128), 0, stream>>>(x16, Wt16, bq, bk, bv, q16, k16, vT16);
    rel_attn<<<dim3(32, BATCH * NHEADS), dim3(128), 0, stream>>>(q16, k16, vT16, Er16, out);
}
